// TransformerBlock_2671469658349
// MI455X (gfx1250) — compile-verified, matrix operand order checked by reading, not run on hardware
//
#include <hip/hip_runtime.h>

// MI455X / gfx1250 transformer block: f16 WMMA (fp32 accumulate) everywhere,
// flash attention, wave32 tiling. Workspace layout uses ~208 MB of d_ws.

typedef _Float16 v16h  __attribute__((ext_vector_type(16)));
typedef _Float16 half8 __attribute__((ext_vector_type(8)));
typedef float    v8f   __attribute__((ext_vector_type(8)));

#define SEQ     2048
#define DMODEL  2048
#define DFF     8192
#define DHEAD   128

// Load one 16xK=32 A-fragment (or N x K=32 B^T-fragment) row for this lane
// from a K-contiguous row of halves. Layout per ISA 7.12.2:
//   lanes 0-15: K = {0..7, 16..23};  lanes 16-31: K = {8..15, 24..31}
static __device__ __forceinline__ v16h ld_frag(const _Float16* p, int lane) {
  const int base = (lane >> 4) << 3;
  half8 lo = *(const half8*)(p + base);
  half8 hi = *(const half8*)(p + base + 16);
  v16h r;
#pragma unroll
  for (int i = 0; i < 8; ++i) { r[i] = lo[i]; r[i + 8] = hi[i]; }
  return r;
}

// ---------------------------------------------------------------------------
// fp32 [K][N] -> f16 [N][K] transpose-convert (32x32 LDS tile)
// ---------------------------------------------------------------------------
__global__ __launch_bounds__(256)
void tconv_kernel(const float* __restrict__ src, _Float16* __restrict__ dst,
                  int K, int N) {
  __shared__ float tile[32][33];
  const int tx = threadIdx.x, ty = threadIdx.y;   // blockDim = (32, 8)
  const int n0 = blockIdx.x * 32, k0 = blockIdx.y * 32;
#pragma unroll
  for (int i = 0; i < 4; ++i)
    tile[ty + i * 8][tx] = src[(size_t)(k0 + ty + i * 8) * N + n0 + tx];
  __syncthreads();
#pragma unroll
  for (int i = 0; i < 4; ++i)
    dst[(size_t)(n0 + ty + i * 8) * K + k0 + tx] = (_Float16)tile[tx][ty + i * 8];
}

// ---------------------------------------------------------------------------
// LayerNorm over 2048 columns; one block per row; writes f16 (+ optional f32)
// ---------------------------------------------------------------------------
template <bool W32>
__global__ __launch_bounds__(256)
void ln_kernel(const float* __restrict__ x, const float* __restrict__ g,
               const float* __restrict__ b, _Float16* __restrict__ outH,
               float* __restrict__ outF) {
  __shared__ float red[16];
  const int row = blockIdx.x;
  const int tid = threadIdx.x;
  const float* xr = x + (size_t)row * DMODEL;
  float v[8], s1 = 0.f, s2 = 0.f;
#pragma unroll
  for (int i = 0; i < 8; ++i) {
    v[i] = xr[tid + i * 256];
    s1 += v[i]; s2 += v[i] * v[i];
  }
#pragma unroll
  for (int off = 16; off >= 1; off >>= 1) {
    s1 += __shfl_xor(s1, off, 32);
    s2 += __shfl_xor(s2, off, 32);
  }
  const int wave = tid >> 5;
  if ((tid & 31) == 0) { red[wave] = s1; red[wave + 8] = s2; }
  __syncthreads();
  if (tid == 0) {
    float t1 = 0.f, t2 = 0.f;
#pragma unroll
    for (int i = 0; i < 8; ++i) { t1 += red[i]; t2 += red[i + 8]; }
    red[0] = t1; red[1] = t2;
  }
  __syncthreads();
  const float mu  = red[0] * (1.0f / DMODEL);
  const float var = red[1] * (1.0f / DMODEL) - mu * mu;
  const float rs  = rsqrtf(var + 1e-5f);
#pragma unroll
  for (int i = 0; i < 8; ++i) {
    const int c = tid + i * 256;
    const float y = (v[i] - mu) * rs * g[c] + b[c];
    outH[(size_t)row * DMODEL + c] = (_Float16)y;
    if (W32) outF[(size_t)row * DMODEL + c] = y;
  }
}

// ---------------------------------------------------------------------------
// Tiled WMMA GEMM: C[M,N] = A[M,K](f16) @ Bt[N,K](f16)^T, fp32 accumulate.
// Block = 256 threads (8 waves), tile 128x128, BK=32. Wave grid 2x4, each
// wave computes 4x2 = 8 WMMA 16x16 C tiles.
// ---------------------------------------------------------------------------
enum { EPI_F16 = 0, EPI_F16T = 1, EPI_PROJ = 2, EPI_GELU = 3, EPI_FINAL = 4 };

template <int EPI>
__global__ __launch_bounds__(256)
void gemm_kernel(const _Float16* __restrict__ A, const _Float16* __restrict__ Bt,
                 const float* __restrict__ bias, const float* __restrict__ res1,
                 const float* __restrict__ res2, _Float16* __restrict__ outH,
                 float* __restrict__ outF, int M, int N, int K) {
  __shared__ __align__(16) _Float16 As[128][32];
  __shared__ __align__(16) _Float16 Bs[128][32];

  const int tid   = threadIdx.x;
  const int lane  = tid & 31;
  const int wave  = tid >> 5;
  const int waveM = wave >> 2;   // 0..1  (64 rows each)
  const int waveN = wave & 3;    // 0..3  (32 cols each)
  const int m0 = blockIdx.y * 128;
  const int n0 = blockIdx.x * 128;
  const int hi = lane >> 4;
  const int ln = lane & 15;

  v8f acc[4][2];
#pragma unroll
  for (int i = 0; i < 4; ++i)
#pragma unroll
    for (int j = 0; j < 2; ++j)
#pragma unroll
      for (int r = 0; r < 8; ++r) acc[i][j][r] = 0.f;

  const int kIters = K >> 5;
  for (int kb = 0; kb < kIters; ++kb) {
    const int kOff = kb << 5;
    __syncthreads();
#pragma unroll
    for (int c = tid; c < 512; c += 256) {
      const int row = c >> 2, off = (c & 3) << 3;
      *(half8*)(&As[row][off]) = *(const half8*)(A  + (size_t)(m0 + row) * K + kOff + off);
      *(half8*)(&Bs[row][off]) = *(const half8*)(Bt + (size_t)(n0 + row) * K + kOff + off);
    }
    if (kb + 1 < kIters) {  // stream next K-tile toward L2 (global_prefetch_b8)
      const int row = tid >> 2, off = (tid & 3) << 3;
      __builtin_prefetch(A  + (size_t)(m0 + row) * K + kOff + 32 + off, 0, 1);
      __builtin_prefetch(Bt + (size_t)(n0 + row) * K + kOff + 32 + off, 0, 1);
    }
    __syncthreads();

    v16h bfr[2];
#pragma unroll
    for (int nt = 0; nt < 2; ++nt)
      bfr[nt] = ld_frag(&Bs[waveN * 32 + nt * 16 + ln][0], lane);
#pragma unroll
    for (int mt = 0; mt < 4; ++mt) {
      v16h afr = ld_frag(&As[waveM * 64 + mt * 16 + ln][0], lane);
#pragma unroll
      for (int nt = 0; nt < 2; ++nt)
        acc[mt][nt] = __builtin_amdgcn_wmma_f32_16x16x32_f16(
            false, afr, false, bfr[nt], (short)0, acc[mt][nt], false, false);
    }
  }

  // Epilogue. C layout: lane holds (row = r + 8*hi, col = ln) of each tile.
#pragma unroll
  for (int mt = 0; mt < 4; ++mt)
#pragma unroll
    for (int nt = 0; nt < 2; ++nt)
#pragma unroll
      for (int r = 0; r < 8; ++r) {
        const int gm = m0 + waveM * 64 + mt * 16 + r + 8 * hi;
        const int gn = n0 + waveN * 32 + nt * 16 + ln;
        const float v = acc[mt][nt][r];
        if (EPI == EPI_F16) {
          outH[(size_t)gm * N + gn] = (_Float16)v;
        } else if (EPI == EPI_F16T) {          // V^T for attention
          outH[(size_t)gn * M + gm] = (_Float16)v;
        } else if (EPI == EPI_PROJ) {          // + bp + x  -> x1 (fp32)
          const size_t idx = (size_t)gm * N + gn;
          outF[idx] = v + bias[gn] + res1[idx];
        } else if (EPI == EPI_GELU) {          // + b1, exact GELU -> f16
          const float t = v + bias[gn];
          outH[(size_t)gm * N + gn] =
              (_Float16)(0.5f * t * (1.0f + erff(t * 0.70710678118654752f)));
        } else {                               // FINAL: + b2 + x1 + g (fp32)
          const size_t idx = (size_t)gm * N + gn;
          outF[idx] = v + bias[gn] + res1[idx] + res2[idx];
        }
      }
}

// ---------------------------------------------------------------------------
// Flash attention: one block per (q-block of 128, head). 8 waves x 16 q rows.
// Q fragments in registers; K / V^T key-blocks (32 keys) staged in LDS.
// ---------------------------------------------------------------------------
__global__ __launch_bounds__(256)
void attn_kernel(const _Float16* __restrict__ Q, const _Float16* __restrict__ Kf,
                 const _Float16* __restrict__ Vt, _Float16* __restrict__ O) {
  __shared__ __align__(16) _Float16 Ks[32][128];   // [key][d]
  __shared__ __align__(16) _Float16 Vs[128][32];   // [d][key]  (from V^T)
  __shared__ __align__(16) _Float16 Ps[8][16][32]; // per-wave P relayout

  const int tid  = threadIdx.x;
  const int lane = tid & 31;
  const int wave = tid >> 5;
  const int qb   = blockIdx.x;
  const int head = blockIdx.y;
  const int hOff = head * DHEAD;
  const int hi   = lane >> 4;
  const int ln   = lane & 15;

  // Q fragments: wave owns q rows [qb*128 + wave*16, +16); 4 K-steps over d=128
  const int qRowFrag = qb * 128 + wave * 16 + ln;
  v16h qf[4];
#pragma unroll
  for (int s = 0; s < 4; ++s)
    qf[s] = ld_frag(Q + (size_t)qRowFrag * DMODEL + hOff + s * 32, lane);

  v8f o[8];
  float m[8], l[8];
#pragma unroll
  for (int dt = 0; dt < 8; ++dt)
#pragma unroll
    for (int r = 0; r < 8; ++r) o[dt][r] = 0.f;
#pragma unroll
  for (int r = 0; r < 8; ++r) { m[r] = -1e30f; l[r] = 0.f; }

  const int kend = qb * 128 + 128;                // causal upper bound
  for (int k0 = 0; k0 < kend; k0 += 32) {
    __syncthreads();
#pragma unroll
    for (int c = tid; c < 512; c += 256) {
      { const int row = c >> 4, off = (c & 15) << 3;   // Ks: 32 x 16 chunks
        *(half8*)(&Ks[row][off]) =
            *(const half8*)(Kf + (size_t)(k0 + row) * DMODEL + hOff + off); }
      { const int row = c >> 2, off = (c & 3) << 3;    // Vs: 128 x 4 chunks
        *(half8*)(&Vs[row][off]) =
            *(const half8*)(Vt + (size_t)(hOff + row) * SEQ + k0 + off); }
    }
    __syncthreads();

    // scores S = Q K^T / sqrt(dhead): 16 q x 32 keys = 2 C tiles, 4 K-steps
    v8f sc[2];
#pragma unroll
    for (int ct = 0; ct < 2; ++ct) {
#pragma unroll
      for (int r = 0; r < 8; ++r) sc[ct][r] = 0.f;
#pragma unroll
      for (int s = 0; s < 4; ++s) {
        v16h kfr = ld_frag(&Ks[ct * 16 + ln][s * 32], lane);
        sc[ct] = __builtin_amdgcn_wmma_f32_16x16x32_f16(
            false, qf[s], false, kfr, (short)0, sc[ct], false, false);
      }
    }
    // scale + causal mask
#pragma unroll
    for (int ct = 0; ct < 2; ++ct)
#pragma unroll
      for (int r = 0; r < 8; ++r) {
        const int qrow = qb * 128 + wave * 16 + r + 8 * hi;
        const int krow = k0 + ct * 16 + ln;
        const float sv = sc[ct][r] * 0.088388347648318447f;  // 1/sqrt(128)
        sc[ct][r] = (krow > qrow) ? -1e30f : sv;
      }
    // online softmax; C-tile rows live across the 16 lanes of each half-wave
    float scale[8];
#pragma unroll
    for (int r = 0; r < 8; ++r) {
      float mx = fmaxf(sc[0][r], sc[1][r]);
#pragma unroll
      for (int off = 8; off >= 1; off >>= 1) mx = fmaxf(mx, __shfl_xor(mx, off, 16));
      const float mnew = fmaxf(m[r], mx);
      scale[r] = __expf(m[r] - mnew);
      m[r] = mnew;
      const float p0 = (sc[0][r] < -1e29f) ? 0.f : __expf(sc[0][r] - mnew);
      const float p1 = (sc[1][r] < -1e29f) ? 0.f : __expf(sc[1][r] - mnew);
      sc[0][r] = p0; sc[1][r] = p1;
      float ps = p0 + p1;
#pragma unroll
      for (int off = 8; off >= 1; off >>= 1) ps += __shfl_xor(ps, off, 16);
      l[r] = l[r] * scale[r] + ps;
    }
#pragma unroll
    for (int dt = 0; dt < 8; ++dt)
#pragma unroll
      for (int r = 0; r < 8; ++r) o[dt][r] *= scale[r];

    // P (C layout) -> per-wave LDS -> A fragment (16 q x 32 key)
#pragma unroll
    for (int ct = 0; ct < 2; ++ct)
#pragma unroll
      for (int r = 0; r < 8; ++r)
        Ps[wave][r + 8 * hi][ct * 16 + ln] = (_Float16)sc[ct][r];
    v16h pa = ld_frag(&Ps[wave][ln][0], lane);
#pragma unroll
    for (int dt = 0; dt < 8; ++dt) {
      v16h vf = ld_frag(&Vs[dt * 16 + ln][0], lane);
      o[dt] = __builtin_amdgcn_wmma_f32_16x16x32_f16(
          false, pa, false, vf, (short)0, o[dt], false, false);
    }
  }

  // normalize + store f16 attention output
#pragma unroll
  for (int r = 0; r < 8; ++r) l[r] = 1.0f / l[r];
#pragma unroll
  for (int dt = 0; dt < 8; ++dt)
#pragma unroll
    for (int r = 0; r < 8; ++r) {
      const int qrow = qb * 128 + wave * 16 + r + 8 * hi;
      O[(size_t)qrow * DMODEL + hOff + dt * 16 + ln] = (_Float16)(o[dt][r] * l[r]);
    }
}

// ---------------------------------------------------------------------------
extern "C" void kernel_launch(void* const* d_in, const int* in_sizes, int n_in,
                              void* d_out, int out_size, void* d_ws, size_t ws_size,
                              hipStream_t stream) {
  (void)in_sizes; (void)n_in; (void)out_size; (void)ws_size;
  const float* x    = (const float*)d_in[0];
  const float* ln1g = (const float*)d_in[1];
  const float* ln1b = (const float*)d_in[2];
  const float* Wq   = (const float*)d_in[3];
  const float* Wk   = (const float*)d_in[4];
  const float* Wv   = (const float*)d_in[5];
  const float* Wp   = (const float*)d_in[6];
  const float* bp   = (const float*)d_in[7];
  const float* ln2g = (const float*)d_in[8];
  const float* ln2b = (const float*)d_in[9];
  const float* W1   = (const float*)d_in[10];
  const float* b1   = (const float*)d_in[11];
  const float* W2   = (const float*)d_in[12];
  const float* b2   = (const float*)d_in[13];
  float* out = (float*)d_out;

  char* ws = (char*)d_ws;
  const size_t MB = (size_t)1 << 20;
  _Float16* WqT = (_Float16*)(ws +   0 * MB);  //  8 MB
  _Float16* WkT = (_Float16*)(ws +   8 * MB);  //  8 MB
  _Float16* WvT = (_Float16*)(ws +  16 * MB);  //  8 MB
  _Float16* WpT = (_Float16*)(ws +  24 * MB);  //  8 MB
  _Float16* W1T = (_Float16*)(ws +  32 * MB);  // 32 MB
  _Float16* W2T = (_Float16*)(ws +  64 * MB);  // 32 MB
  _Float16* hF  = (_Float16*)(ws +  96 * MB);  //  8 MB  LN1 out
  _Float16* Qf  = (_Float16*)(ws + 104 * MB);  //  8 MB
  _Float16* Kf  = (_Float16*)(ws + 112 * MB);  //  8 MB
  _Float16* Vt  = (_Float16*)(ws + 120 * MB);  //  8 MB  V transposed
  _Float16* At  = (_Float16*)(ws + 128 * MB);  //  8 MB  attn out
  float*    x1  = (float*)  (ws + 136 * MB);   // 16 MB
  float*    g32 = (float*)  (ws + 152 * MB);   // 16 MB  LN2 out (fp32)
  _Float16* g16 = (_Float16*)(ws + 168 * MB);  //  8 MB  LN2 out (f16)
  _Float16* h1  = (_Float16*)(ws + 176 * MB);  // 32 MB  MLP hidden

  const dim3 tb(32, 8);
  tconv_kernel<<<dim3( 64,  64), tb, 0, stream>>>(Wq, WqT, DMODEL, DMODEL);
  tconv_kernel<<<dim3( 64,  64), tb, 0, stream>>>(Wk, WkT, DMODEL, DMODEL);
  tconv_kernel<<<dim3( 64,  64), tb, 0, stream>>>(Wv, WvT, DMODEL, DMODEL);
  tconv_kernel<<<dim3( 64,  64), tb, 0, stream>>>(Wp, WpT, DMODEL, DMODEL);
  tconv_kernel<<<dim3(256,  64), tb, 0, stream>>>(W1, W1T, DMODEL, DFF);
  tconv_kernel<<<dim3( 64, 256), tb, 0, stream>>>(W2, W2T, DFF, DMODEL);

  ln_kernel<false><<<SEQ, 256, 0, stream>>>(x, ln1g, ln1b, hF, nullptr);

  gemm_kernel<EPI_F16 ><<<dim3(16, 16), 256, 0, stream>>>(
      hF, WqT, nullptr, nullptr, nullptr, Qf, nullptr, SEQ, DMODEL, DMODEL);
  gemm_kernel<EPI_F16 ><<<dim3(16, 16), 256, 0, stream>>>(
      hF, WkT, nullptr, nullptr, nullptr, Kf, nullptr, SEQ, DMODEL, DMODEL);
  gemm_kernel<EPI_F16T><<<dim3(16, 16), 256, 0, stream>>>(
      hF, WvT, nullptr, nullptr, nullptr, Vt, nullptr, SEQ, DMODEL, DMODEL);

  attn_kernel<<<dim3(16, 16), 256, 0, stream>>>(Qf, Kf, Vt, At);

  gemm_kernel<EPI_PROJ><<<dim3(16, 16), 256, 0, stream>>>(
      At, WpT, bp, x, nullptr, nullptr, x1, SEQ, DMODEL, DMODEL);

  ln_kernel<true><<<SEQ, 256, 0, stream>>>(x1, ln2g, ln2b, g16, g32);

  gemm_kernel<EPI_GELU><<<dim3(64, 16), 256, 0, stream>>>(
      g16, W1T, b1, nullptr, nullptr, h1, nullptr, SEQ, DFF, DMODEL);
  gemm_kernel<EPI_FINAL><<<dim3(16, 16), 256, 0, stream>>>(
      h1, W2T, b2, x1, g32, nullptr, out, SEQ, DMODEL, DFF);
}
